// PatchTransformer_15857019257409
// MI455X (gfx1250) — matrix operand order repaired and checked by reading, not compiled, not run on hardware
//
#include <hip/hip_runtime.h>
#include <hip/hip_bf16.h>

// ---------------------------------------------------------------------------
// PatchTransformer for MI455X (gfx1250): wave32 + WMMA f32_16x16x32_f16
// Pipeline: [weight prep fp32->fp16 transposed] -> [KNN top-8 via LDS point
// cloud] -> [pos MLP, 2 WMMA GEMMs + per-fragment max-reduce] -> [fused token
// transformer: 6 chained WMMA GEMMs through LDS with fp32 softmax].
// ---------------------------------------------------------------------------

typedef __attribute__((ext_vector_type(16))) _Float16 v16h;
typedef __attribute__((ext_vector_type(8)))  _Float16 v8h;
typedef __attribute__((ext_vector_type(8)))  float    v8f;

#define BATCH 4
#define SEQ   4096
#define DIM   256
#define NTOK  (BATCH * SEQ)      // 16384
#define KNN   8
#define XROW  259                // 3 + 256

__device__ inline v8f zero8() {
  v8f z = {0.f, 0.f, 0.f, 0.f, 0.f, 0.f, 0.f, 0.f};
  return z;
}

__device__ inline float gelu_exact(float x) {
  return 0.5f * x * (1.0f + erff(x * 0.70710678118654752f));
}

__device__ inline v8f wmma32(v16h a, v16h b, v8f c) {
  // v_wmma_f32_16x16x32_f16: D = A(16x32) * B(32x16) + C(16x16 f32)
  return __builtin_amdgcn_wmma_f32_16x16x32_f16(
      false, a, false, b, (short)0, c, false, false);
}

// A fragment (16-bit A, 16x32 MxK; ISA 7.12.2):
// lanes 0-15 : row M = m0+lane,     K = k0+{0..7} (v0..3),  k0+{16..23} (v4..7)
// lanes 16-31: row M = m0+lane-16,  K = k0+{8..15},         k0+{24..31}
__device__ inline v16h loadA(const _Float16* A, int lda, int m0, int k0) {
  int lane = threadIdx.x & 31;
  int r = lane & 15, hi = lane >> 4;
  const _Float16* p = A + (size_t)(m0 + r) * lda + k0 + hi * 8;
  union { v16h v; v8h h[2]; } u;
  u.h[0] = *(const v8h*)(p);
  u.h[1] = *(const v8h*)(p + 16);
  return u.v;
}

// B fragment (16-bit B, 32x16 KxN), weights stored transposed (N x K row-major):
// lanes 0-15 : col N = n0+lane,     K = k0+{0..7},  k0+{16..23}
// lanes 16-31: col N = n0+lane-16,  K = k0+{8..15}, k0+{24..31}
__device__ inline v16h loadB(const _Float16* wT, int ldb, int n0, int k0) {
  int lane = threadIdx.x & 31;
  int r = lane & 15, hi = lane >> 4;
  const _Float16* p = wT + (size_t)(n0 + r) * ldb + k0 + hi * 8;
  union { v16h v; v8h h[2]; } u;
  u.h[0] = *(const v8h*)(p);
  u.h[1] = *(const v8h*)(p + 16);
  return u.v;
}

// K-loop accumulate over NT n-tiles for one wave (m0 fixed, n = n0base + t*16)
template <int NT>
__device__ inline void gemm_acc(const _Float16* A, int lda, const _Float16* wT,
                                int K, int m0, int n0base, v8f* acc) {
  for (int k0 = 0; k0 < K; k0 += 32) {
    v16h a = loadA(A, lda, m0, k0);
#pragma unroll
    for (int t = 0; t < NT; ++t) {
      v16h b = loadB(wT, K, n0base + t * 16, k0);
      acc[t] = wmma32(a, b, acc[t]);
    }
  }
}

// ---------------------------------------------------------------------------
// Weight prep: fp32 (K x N) -> fp16 transposed (N x K), optional subtraction
// ---------------------------------------------------------------------------
__global__ __launch_bounds__(256) void convertT_kernel(const float* __restrict__ src,
                                                       const float* __restrict__ sub,
                                                       _Float16* __restrict__ dst,
                                                       int K, int N) {
  int i = blockIdx.x * 256 + threadIdx.x;
  if (i >= K * N) return;
  int n = i / K, k = i - n * K;
  float v = src[(size_t)k * N + n];
  if (sub) v -= sub[(size_t)k * N + n];
  dst[(size_t)n * K + k] = (_Float16)v;
}

// Wp1 (10 x 128) -> fp16 transposed + K padded to 32: (128 x 32)
__global__ __launch_bounds__(256) void convWp1_kernel(const float* __restrict__ wp1,
                                                      _Float16* __restrict__ dst) {
  int i = blockIdx.x * 256 + threadIdx.x;
  if (i >= 128 * 32) return;
  int n = i >> 5, k = i & 31;
  dst[n * 32 + k] = (_Float16)(k < 10 ? wp1[k * 128 + n] : 0.f);
}

// ---------------------------------------------------------------------------
// KNN: per-batch point cloud in LDS; each thread scans 4096 candidates,
// keeps 8 smallest (stable ties), then writes the 10-dim pos features
// (padded to 32 halves) for the pos-MLP GEMM.
// ---------------------------------------------------------------------------
__global__ __launch_bounds__(256) void knn_kernel(const float* __restrict__ x,
                                                  _Float16* __restrict__ posfeat) {
  __shared__ float cx[SEQ], cy[SEQ], cz[SEQ];
  int b = blockIdx.x >> 4;             // 16 blocks per batch
  int q0 = (blockIdx.x & 15) * 256;
  const float* xb = x + (size_t)b * SEQ * XROW;
  for (int j = threadIdx.x; j < SEQ; j += 256) {
    cx[j] = xb[(size_t)j * XROW + 0];
    cy[j] = xb[(size_t)j * XROW + 1];
    cz[j] = xb[(size_t)j * XROW + 2];
  }
  __syncthreads();

  int q = q0 + threadIdx.x;
  float qx = cx[q], qy = cy[q], qz = cz[q];

  float bd[KNN];
  int   bi[KNN];
#pragma unroll
  for (int k = 0; k < KNN; ++k) { bd[k] = 3.4e38f; bi[k] = 0; }

  for (int j = 0; j < SEQ; ++j) {
    float dx = cx[j] - qx, dy = cy[j] - qy, dz = cz[j] - qz;
    float d = dx * dx + dy * dy + dz * dz;
    if (d < bd[KNN - 1]) {             // strict <: stable tie-break by index
      bd[KNN - 1] = d; bi[KNN - 1] = j;
#pragma unroll
      for (int t = KNN - 1; t > 0; --t) {
        if (bd[t] < bd[t - 1]) {
          float td = bd[t]; bd[t] = bd[t - 1]; bd[t - 1] = td;
          int ti = bi[t]; bi[t] = bi[t - 1]; bi[t - 1] = ti;
        }
      }
    }
  }

  size_t row = ((size_t)(b * SEQ + q)) * KNN;
#pragma unroll
  for (int k = 0; k < KNN; ++k) {
    int j = bi[k];
    float kx = cx[j], ky = cy[j], kz = cz[j];
    float rx = kx - qx, ry = ky - qy, rz = kz - qz;
    float nrm = sqrtf(rx * rx + ry * ry + rz * rz);
    _Float16* f = posfeat + (row + k) * 32;
    f[0] = (_Float16)qx; f[1] = (_Float16)qy; f[2] = (_Float16)qz;
    f[3] = (_Float16)kx; f[4] = (_Float16)ky; f[5] = (_Float16)kz;
    f[6] = (_Float16)rx; f[7] = (_Float16)ry; f[8] = (_Float16)rz;
    f[9] = (_Float16)nrm;
#pragma unroll
    for (int t = 10; t < 32; ++t) f[t] = (_Float16)0.f;
  }
}

// ---------------------------------------------------------------------------
// Pos MLP: block = 16 tokens = 128 feature rows.
// GEMM1: (128x32 pad) @ Wp1T -> gelu -> hid LDS (128x128 f16)
// GEMM2: (128x128) @ Wp2T -> 128x256; m-tile of 16 rows = 2 tokens, so each
// lane's 8 C VGPRs are exactly one token's 8 k-rows -> per-lane max = pos_emb.
// ---------------------------------------------------------------------------
__global__ __launch_bounds__(256) void pos_kernel(const _Float16* __restrict__ posfeat,
                                                  const _Float16* __restrict__ wP1T,
                                                  const _Float16* __restrict__ wP2T,
                                                  const float* __restrict__ bp1,
                                                  const float* __restrict__ bp2,
                                                  float* __restrict__ posEmb) {
  __shared__ _Float16 hid[128 * 128];
  int tid = threadIdx.x, wave = tid >> 5, lane = tid & 31;
  int r = lane & 15, hi = lane >> 4;
  const _Float16* A = posfeat + (size_t)blockIdx.x * 128 * 32;
  int m0 = wave * 16;

  // GEMM1: K=32 (single WMMA per tile), N=128 (8 n-tiles per wave)
  v8f acc[8];
#pragma unroll
  for (int t = 0; t < 8; ++t) acc[t] = zero8();
  gemm_acc<8>(A, 32, wP1T, 32, m0, 0, acc);
#pragma unroll
  for (int t = 0; t < 8; ++t) {
    int col = t * 16 + r;
#pragma unroll
    for (int j = 0; j < 8; ++j) {
      int m = m0 + hi * 8 + j;
      hid[m * 128 + col] = (_Float16)gelu_exact(acc[t][j] + bp1[col]);
    }
  }
  __syncthreads();

  // GEMM2: K=128, N=256 (16 n-tiles per wave), then max over k (8 VGPRs)
  v8f acc2[16];
#pragma unroll
  for (int t = 0; t < 16; ++t) acc2[t] = zero8();
  gemm_acc<16>(hid, 128, wP2T, 128, m0, 0, acc2);

  int tokBase = blockIdx.x * 16;
#pragma unroll
  for (int t = 0; t < 16; ++t) {
    int col = t * 16 + r;
    float mx = acc2[t][0];
#pragma unroll
    for (int j = 1; j < 8; ++j) mx = fmaxf(mx, acc2[t][j]);
    int token = tokBase + (m0 >> 3) + hi;   // rows m0..m0+15 = tokens 2w,2w+1
    posEmb[(size_t)token * DIM + col] = mx + bp2[col];
  }
}

// ---------------------------------------------------------------------------
// Fused token transformer: 64-token tile, 6 WMMA GEMMs chained through LDS.
// Dynamic LDS 128KB: aL(32K f16) | vL(32K f16) | region(64K: hL f16 aliases
// the fp32 softmax scratch sL, which only becomes live after hL is dead).
// q-k == h @ (Wq - Wk)  -> wQKT precomputed, saves a full GEMM.
// ---------------------------------------------------------------------------
__global__ __launch_bounds__(256) void main_kernel(
    const float* __restrict__ x, const float* __restrict__ bIn,
    const float* __restrict__ bg1, const float* __restrict__ bg2,
    const float* __restrict__ bOut, const _Float16* __restrict__ wInT,
    const _Float16* __restrict__ wQKT, const _Float16* __restrict__ wVT,
    const _Float16* __restrict__ wG1T, const _Float16* __restrict__ wG2T,
    const _Float16* __restrict__ wOutT, const float* __restrict__ posEmb,
    float* __restrict__ out) {
  extern __shared__ char smem[];
  _Float16* aL = (_Float16*)smem;                    // 64x256 f16
  _Float16* vL = (_Float16*)(smem + 32768);          // 64x256 f16
  float*    sL = (float*)(smem + 65536);             // 64x256 f32 (phase 4+)
  _Float16* hL = (_Float16*)(smem + 65536);          // 64x256 f16 (phase 1-2)

  const int MT = 64;
  int tok0 = blockIdx.x * MT;
  int tid = threadIdx.x, wave = tid >> 5, lane = tid & 31;
  int r = lane & 15, hi = lane >> 4;
  int m0 = (wave & 3) * 16;
  int nBase = (wave >> 2) * 128;

  // pre -> aL (f16)
  for (int i = tid; i < MT * DIM; i += 256) {
    int m = i >> 8, c = i & 255;
    aL[m * DIM + c] = (_Float16)x[(size_t)(tok0 + m) * XROW + 3 + c];
  }
  __syncthreads();

  v8f acc[8];

  // Phase 1: h = pre @ W_in + b_in + pos_emb  -> hL
#pragma unroll
  for (int t = 0; t < 8; ++t) acc[t] = zero8();
  gemm_acc<8>(aL, DIM, wInT, DIM, m0, nBase, acc);
#pragma unroll
  for (int t = 0; t < 8; ++t) {
    int col = nBase + t * 16 + r;
#pragma unroll
    for (int j = 0; j < 8; ++j) {
      int m = m0 + hi * 8 + j;
      float v = acc[t][j] + bIn[col] + posEmb[(size_t)(tok0 + m) * DIM + col];
      hL[m * DIM + col] = (_Float16)v;
    }
  }
  __syncthreads();

  // Phase 2a: v = h @ Wv -> vL (f16)
#pragma unroll
  for (int t = 0; t < 8; ++t) acc[t] = zero8();
  gemm_acc<8>(hL, DIM, wVT, DIM, m0, nBase, acc);
#pragma unroll
  for (int t = 0; t < 8; ++t) {
    int col = nBase + t * 16 + r;
#pragma unroll
    for (int j = 0; j < 8; ++j)
      vL[(m0 + hi * 8 + j) * DIM + col] = (_Float16)acc[t][j];
  }
  // Phase 2b: t = h @ (Wq - Wk) -> aL (f16)
#pragma unroll
  for (int t = 0; t < 8; ++t) acc[t] = zero8();
  gemm_acc<8>(hL, DIM, wQKT, DIM, m0, nBase, acc);
#pragma unroll
  for (int t = 0; t < 8; ++t) {
    int col = nBase + t * 16 + r;
#pragma unroll
    for (int j = 0; j < 8; ++j)
      aL[(m0 + hi * 8 + j) * DIM + col] = (_Float16)acc[t][j];
  }
  __syncthreads();

  // Phase 3: g = gelu(t @ Wg1 + bg1) -> aL in place (read-all, sync, write)
#pragma unroll
  for (int t = 0; t < 8; ++t) acc[t] = zero8();
  gemm_acc<8>(aL, DIM, wG1T, DIM, m0, nBase, acc);
  __syncthreads();
#pragma unroll
  for (int t = 0; t < 8; ++t) {
    int col = nBase + t * 16 + r;
#pragma unroll
    for (int j = 0; j < 8; ++j)
      aL[(m0 + hi * 8 + j) * DIM + col] =
          (_Float16)gelu_exact(acc[t][j] + bg1[col]);
  }
  __syncthreads();

  // Phase 4: logits = (g @ Wg2 + bg2) * 256^-0.5 -> sL (f32, overlays dead hL)
#pragma unroll
  for (int t = 0; t < 8; ++t) acc[t] = zero8();
  gemm_acc<8>(aL, DIM, wG2T, DIM, m0, nBase, acc);
#pragma unroll
  for (int t = 0; t < 8; ++t) {
    int col = nBase + t * 16 + r;
#pragma unroll
    for (int j = 0; j < 8; ++j)
      sL[(m0 + hi * 8 + j) * DIM + col] = (acc[t][j] + bg2[col]) * 0.0625f;
  }
  __syncthreads();

  // Phase 5: row softmax (256 logits, fp32) * v  -> aL (f16)
  if (tid < MT) {
    int rr = tid;
    float mx = -3.4e38f;
    for (int n = 0; n < DIM; ++n) mx = fmaxf(mx, sL[rr * DIM + n]);
    float sum = 0.f;
    for (int n = 0; n < DIM; ++n) {
      float e = __expf(sL[rr * DIM + n] - mx);
      sL[rr * DIM + n] = e;
      sum += e;
    }
    float inv = 1.f / sum;
    for (int n = 0; n < DIM; ++n)
      aL[rr * DIM + n] =
          (_Float16)(sL[rr * DIM + n] * inv * (float)vL[rr * DIM + n]);
  }
  __syncthreads();

  // Phase 6: out = res @ W_out + b_out + pre  (fp32 store)
#pragma unroll
  for (int t = 0; t < 8; ++t) acc[t] = zero8();
  gemm_acc<8>(aL, DIM, wOutT, DIM, m0, nBase, acc);
#pragma unroll
  for (int t = 0; t < 8; ++t) {
    int col = nBase + t * 16 + r;
#pragma unroll
    for (int j = 0; j < 8; ++j) {
      int m = m0 + hi * 8 + j;
      out[(size_t)(tok0 + m) * DIM + col] =
          acc[t][j] + bOut[col] + x[(size_t)(tok0 + m) * XROW + 3 + col];
    }
  }
}

// ---------------------------------------------------------------------------
extern "C" void kernel_launch(void* const* d_in, const int* in_sizes, int n_in,
                              void* d_out, int out_size, void* d_ws,
                              size_t ws_size, hipStream_t stream) {
  const float* x     = (const float*)d_in[0];
  const float* W_in  = (const float*)d_in[1];
  const float* b_in  = (const float*)d_in[2];
  const float* Wq    = (const float*)d_in[3];
  const float* Wk    = (const float*)d_in[4];
  const float* Wv    = (const float*)d_in[5];
  const float* Wg1   = (const float*)d_in[6];
  const float* bg1   = (const float*)d_in[7];
  const float* Wg2   = (const float*)d_in[8];
  const float* bg2   = (const float*)d_in[9];
  const float* W_out = (const float*)d_in[10];
  const float* b_out = (const float*)d_in[11];
  const float* Wp1   = (const float*)d_in[12];
  const float* bp1   = (const float*)d_in[13];
  const float* Wp2   = (const float*)d_in[14];
  const float* bp2   = (const float*)d_in[15];

  // Workspace layout (256B-aligned segments)
  size_t off = 0;
  auto alloc = [&](size_t bytes) {
    void* p = (char*)d_ws + off;
    off += (bytes + 255) & ~(size_t)255;
    return p;
  };
  _Float16* wInT    = (_Float16*)alloc(256 * 256 * 2);
  _Float16* wQKT    = (_Float16*)alloc(256 * 256 * 2);
  _Float16* wVT     = (_Float16*)alloc(256 * 256 * 2);
  _Float16* wG1T    = (_Float16*)alloc(256 * 256 * 2);
  _Float16* wG2T    = (_Float16*)alloc(256 * 256 * 2);
  _Float16* wOutT   = (_Float16*)alloc(256 * 256 * 2);
  _Float16* wP1T    = (_Float16*)alloc(128 * 32 * 2);
  _Float16* wP2T    = (_Float16*)alloc(256 * 128 * 2);
  _Float16* posfeat = (_Float16*)alloc((size_t)NTOK * KNN * 32 * 2);  // 8 MB
  float*    posEmb  = (float*)alloc((size_t)NTOK * DIM * 4);          // 16 MB

  // Weight prep (all fp16, transposed N x K)
  int g64k = (256 * 256 + 255) / 256;
  convertT_kernel<<<g64k, 256, 0, stream>>>(W_in, nullptr, wInT, 256, 256);
  convertT_kernel<<<g64k, 256, 0, stream>>>(Wq, Wk, wQKT, 256, 256);  // Wq-Wk
  convertT_kernel<<<g64k, 256, 0, stream>>>(Wv, nullptr, wVT, 256, 256);
  convertT_kernel<<<g64k, 256, 0, stream>>>(Wg1, nullptr, wG1T, 256, 256);
  convertT_kernel<<<g64k, 256, 0, stream>>>(Wg2, nullptr, wG2T, 256, 256);
  convertT_kernel<<<g64k, 256, 0, stream>>>(W_out, nullptr, wOutT, 256, 256);
  convWp1_kernel<<<16, 256, 0, stream>>>(Wp1, wP1T);
  convertT_kernel<<<(128 * 256 + 255) / 256, 256, 0, stream>>>(
      Wp2, nullptr, wP2T, 128, 256);

  // KNN + pos features (64 blocks: 16 per batch x 256 queries each)
  knn_kernel<<<BATCH * (SEQ / 256), 256, 0, stream>>>(x, posfeat);

  // Pos MLP (16 tokens / block)
  pos_kernel<<<NTOK / 16, 256, 0, stream>>>(posfeat, wP1T, wP2T, bp1, bp2,
                                            posEmb);

  // Fused transformer (64 tokens / block, 128 KB dynamic LDS)
  const int SMEM = 131072;
  (void)hipFuncSetAttribute((const void*)main_kernel,
                            hipFuncAttributeMaxDynamicSharedMemorySize, SMEM);
  main_kernel<<<NTOK / 64, 256, SMEM, stream>>>(
      x, b_in, bg1, bg2, b_out, wInT, wQKT, wVT, wG1T, wG2T, wOutT, posEmb,
      (float*)d_out);
}
